// RNN_43078521979535
// MI455X (gfx1250) — matrix operand order repaired and checked by reading, not compiled, not run on hardware
//
#include <hip/hip_runtime.h>
#include <hip/hip_bf16.h>

#define SEQ_LEN 4096
#define IN_DIM  256
#define HID     512
#define WI_LD   (IN_DIM + HID)   // 768
#define OUT_DIM 5
#define CHUNK   64
#define NCHUNK  64               // SEQ_LEN / CHUNK
#define MAT_ELEMS (HID * HID)    // 262144 elements per 512x512 matrix
#define MAT_BYTES ((size_t)MAT_ELEMS * 2)

typedef __attribute__((ext_vector_type(16))) __bf16 bf16x16;
typedef __attribute__((ext_vector_type(8)))  __bf16 bf16x8;
typedef __attribute__((ext_vector_type(8)))  float  f32x8;
typedef __attribute__((ext_vector_type(4)))  unsigned int u32x4;
typedef __attribute__((ext_vector_type(8)))  int i32x8;
typedef __attribute__((ext_vector_type(4)))  int i32x4;

union FragU { bf16x16 v; bf16x8 h[2]; __bf16 e[16]; };

// ---------------------------------------------------------------------------
// WMMA helpers (CDNA5 16x16x32 bf16, f32 accum)
// A fragment (16x32): lane L<16 holds row M=L, K = {k0..k0+7, k0+16..k0+23};
//                     lane L>=16 holds row M=L-16, K shifted by +8.
// B fragment (32x16) from W (N x K row-major), B[k][n] = W[n][k]:
//                     lane L<16 holds col N=L, K = {k0..k0+7, k0+16..k0+23};
//                     lane L>=16 holds col N=L-16, K shifted by +8.
// C/D (16x16 f32):    VGPR r, lane L -> (M = r + (L>=16 ? 8:0), N = L%16)
// ---------------------------------------------------------------------------
__device__ __forceinline__ bf16x16 load_frag_a(const __bf16* A, int ld, int row0, int k0) {
  int lane = threadIdx.x & 31;
  const __bf16* p = A + (size_t)(row0 + (lane & 15)) * ld + k0 + ((lane >> 4) << 3);
  FragU f;
  f.h[0] = *(const bf16x8*)(p);
  f.h[1] = *(const bf16x8*)(p + 16);
  return f.v;
}

__device__ __forceinline__ bf16x16 load_frag_bT(const __bf16* W, int ld, int n0, int k0) {
  int lane = threadIdx.x & 31;
  const __bf16* p = W + (size_t)(n0 + (lane & 15)) * ld + k0 + ((lane >> 4) << 3);
  FragU f;
  f.h[0] = *(const bf16x8*)(p);
  f.h[1] = *(const bf16x8*)(p + 16);
  return f.v;
}

__device__ __forceinline__ f32x8 wmma_bf16(bf16x16 a, bf16x16 b, f32x8 c) {
  return __builtin_amdgcn_wmma_f32_16x16x32_bf16(false, a, false, b, (short)0, c,
                                                 false, false);
}

// ---------------------------------------------------------------------------
// TDM: issue an async 2D tile load (tile_w x tile_h, bf16 elements) from a
// row-major global matrix (row stride = row_stride elements) into LDS.
// D# built per CDNA5 ISA 8.3/8.4: group0 {count=1, lds_addr, global_addr,
// type=2}, group1 {data_size=2B, tensor dims (huge, no OOB), tile dims,
// dim0 stride}. Groups 2/3 zero (2D tensor). Tracked by TENSORcnt.
// This toolchain's builtin takes 6 args (extra i32x8 group before cpol).
// ---------------------------------------------------------------------------
__device__ __forceinline__ void tdm_load_tile_2d(const __bf16* gsrc,
                                                 unsigned lds_off,
                                                 unsigned tile_w, unsigned tile_h,
                                                 unsigned row_stride) {
  unsigned long long ga = (unsigned long long)gsrc;
  u32x4 g0;
  g0.x = 1u;                                                 // count=1 (valid user D#)
  g0.y = lds_off;                                            // LDS byte address
  g0.z = (unsigned)ga;                                       // global_addr[31:0]
  g0.w = (unsigned)((ga >> 32) & 0x01FFFFFFu) | (2u << 30);  // addr[56:32] | type=2
  const unsigned td0 = 0x40000000u, td1 = 0x40000000u;       // huge tensor dims
  i32x8 g1;
  g1[0] = (int)(1u << 16);                                   // data_size = 2 bytes
  g1[1] = (int)((td0 & 0xFFFFu) << 16);                      // tensor_dim0[15:0]
  g1[2] = (int)((td0 >> 16) | ((td1 & 0xFFFFu) << 16));      // td0[31:16] | td1[15:0]
  g1[3] = (int)((td1 >> 16) | (tile_w << 16));               // td1[31:16] | tile_dim0
  g1[4] = (int)(tile_h & 0xFFFFu);                           // tile_dim1 (tile_dim2=0)
  g1[5] = (int)row_stride;                                   // tensor_dim0_stride[31:0]
  g1[6] = 0;                                                 // stride hi, dim1_stride lo
  g1[7] = 0;
  i32x4 z4 = {0, 0, 0, 0};
  i32x8 z8 = {0, 0, 0, 0, 0, 0, 0, 0};
  __builtin_amdgcn_tensor_load_to_lds(g0, g1, z4, z4, z8, 0);
}

// ---------------------------------------------------------------------------
// Kernel 1: gather embedding rows + xproj = x @ Wx.T + bi, stored as bf16.
// ---------------------------------------------------------------------------
__global__ __launch_bounds__(256) void k_embed_xproj(
    const int* __restrict__ tokens, const float* __restrict__ emb,
    const float* __restrict__ Wi, const float* __restrict__ bi,
    __bf16* __restrict__ xproj) {
  int wave = (blockIdx.x << 3) + (threadIdx.x >> 5);
  int m0 = (wave >> 5) << 4;
  int n0 = (wave & 31) << 4;
  int lane = threadIdx.x & 31;
  int m = m0 + (lane & 15);
  int n = n0 + (lane & 15);
  const float* arow = emb + (size_t)tokens[m] * IN_DIM;
  const float* wrow = Wi + (size_t)n * WI_LD;
  int koffA = (lane >> 4) << 3;
  int koffB = (lane >> 4) << 3;
  f32x8 acc = {0.f, 0.f, 0.f, 0.f, 0.f, 0.f, 0.f, 0.f};
#pragma unroll
  for (int kb = 0; kb < IN_DIM; kb += 32) {
    f32x8 a0 = *(const f32x8*)(arow + kb + koffA);
    f32x8 a1 = *(const f32x8*)(arow + kb + koffA + 16);
    f32x8 b0 = *(const f32x8*)(wrow + kb + koffB);
    f32x8 b1 = *(const f32x8*)(wrow + kb + koffB + 16);
    FragU fa, fb;
#pragma unroll
    for (int i = 0; i < 8; ++i) {
      fa.e[i] = (__bf16)a0[i]; fa.e[8 + i] = (__bf16)a1[i];
      fb.e[i] = (__bf16)b0[i]; fb.e[8 + i] = (__bf16)b1[i];
    }
    acc = wmma_bf16(fa.v, fb.v, acc);
  }
  float bias = bi[n];
  int mbase = m0 + ((lane >> 4) << 3);
#pragma unroll
  for (int r = 0; r < 8; ++r)
    xproj[(size_t)(mbase + r) * HID + n] = (__bf16)(acc[r] + bias);
}

// ---------------------------------------------------------------------------
// Kernel 2: pw[0] = I, pw[1] = bf16(Wh); transposed copies in pwT.
// ---------------------------------------------------------------------------
__global__ __launch_bounds__(256) void k_init_pows(
    const float* __restrict__ Wi, __bf16* __restrict__ pw, __bf16* __restrict__ pwT) {
  int idx = blockIdx.x * 256 + threadIdx.x;
  int r = idx >> 9, c = idx & 511;
  __bf16 id = (__bf16)((r == c) ? 1.0f : 0.0f);
  pw[idx] = id;
  pwT[idx] = id;
  __bf16 w = (__bf16)Wi[(size_t)r * WI_LD + IN_DIM + c];
  pw[MAT_ELEMS + idx] = w;
  pwT[MAT_ELEMS + ((size_t)c << 9) + r] = w;
}

// ---------------------------------------------------------------------------
// Kernel 3: batched power doubling. P_{have+i} = P_have @ P_i, i = 1..count.
// ---------------------------------------------------------------------------
__global__ __launch_bounds__(256) void k_power_gemm(
    __bf16* pw, __bf16* pwT, int have) {
  int i = (int)blockIdx.y + 1;
  int s = have + i;
  const __bf16* A = pw + (size_t)have * MAT_ELEMS;
  const __bf16* W = pwT + (size_t)i * MAT_ELEMS;
  __bf16* D = pw + (size_t)s * MAT_ELEMS;
  int wave = (blockIdx.x << 3) + (threadIdx.x >> 5);
  int m0 = (wave >> 5) << 4;
  int n0 = (wave & 31) << 4;
  f32x8 acc = {0.f, 0.f, 0.f, 0.f, 0.f, 0.f, 0.f, 0.f};
#pragma unroll 4
  for (int kb = 0; kb < HID; kb += 32) {
    if (kb + 32 < HID) {   // L2 prefetch of next K slice (global_prefetch_b8)
      __builtin_prefetch(A + (size_t)m0 * HID + kb + 32, 0, 0);
      __builtin_prefetch(W + (size_t)n0 * HID + kb + 32, 0, 0);
    }
    acc = wmma_bf16(load_frag_a(A, HID, m0, kb), load_frag_bT(W, HID, n0, kb), acc);
  }
  int lane = threadIdx.x & 31;
  int n = n0 + (lane & 15);
  int mbase = m0 + ((lane >> 4) << 3);
#pragma unroll
  for (int r = 0; r < 8; ++r)
    D[(size_t)(mbase + r) * HID + n] = (__bf16)acc[r];
  if (s <= 32) {
    union { bf16x8 v; __bf16 e[8]; } t;
#pragma unroll
    for (int r = 0; r < 8; ++r) t.e[r] = (__bf16)acc[r];
    *(bf16x8*)(pwT + (size_t)s * MAT_ELEMS + (size_t)n * HID + mbase) = t.v;
  }
}

// ---------------------------------------------------------------------------
// Kernel 4: Y(64x512 f32) = xproj.reshape(64, 32768) @ [P63 ... P0]^T.
// TDM double-buffered: per stage s (j = s/2, half = s%2), DMA a 64x256 tile of
// A (Xstack) and of B (P_{63-j} rows n0..n0+63) into LDS; waves compute WMMA
// from LDS. One wave issues tensor_load_to_lds; TENSORcnt + barrier sync.
// 8 blocks x 64 N-columns; 8 waves: 4 M-tiles x 2 N-pairs, 2 accs per wave.
// ---------------------------------------------------------------------------
__global__ __launch_bounds__(256) void k_chunk_gemm(
    const __bf16* __restrict__ xproj, const __bf16* __restrict__ pw,
    float* __restrict__ Y) {
  __shared__ __bf16 sA[2][CHUNK * 256];   // 2 x 32 KiB
  __shared__ __bf16 sB[2][CHUNK * 256];   // 2 x 32 KiB
  const int NSTAGE = 2 * NCHUNK;          // 128 stages of K=256
  int n0g = blockIdx.x << 6;              // block's N origin (0..448)
  int wave = threadIdx.x >> 5;
  int m0 = (wave & 3) << 4;
  int nl0 = (wave >> 2) << 5;             // 0 or 32 (two 16-wide tiles each)
  bool issuer = (wave == 0);

  unsigned ldsA[2], ldsB[2];
  ldsA[0] = (unsigned)(unsigned long long)&sA[0][0];
  ldsA[1] = (unsigned)(unsigned long long)&sA[1][0];
  ldsB[0] = (unsigned)(unsigned long long)&sB[0][0];
  ldsB[1] = (unsigned)(unsigned long long)&sB[1][0];

  f32x8 acc0 = {0.f, 0.f, 0.f, 0.f, 0.f, 0.f, 0.f, 0.f};
  f32x8 acc1 = {0.f, 0.f, 0.f, 0.f, 0.f, 0.f, 0.f, 0.f};

  if (issuer) {  // prologue: stage 0
    tdm_load_tile_2d(xproj, ldsA[0], 256, CHUNK, CHUNK * HID);
    tdm_load_tile_2d(pw + (size_t)63 * MAT_ELEMS + (size_t)n0g * HID,
                     ldsB[0], 256, CHUNK, HID);
  }
  for (int s = 0; s < NSTAGE; ++s) {
    int buf = s & 1;
    if (issuer) {
      if (s + 1 < NSTAGE) {
        int t = s + 1, j = t >> 1, ko = (t & 1) << 8;
        tdm_load_tile_2d(xproj + (size_t)j * HID + ko, ldsA[t & 1],
                         256, CHUNK, CHUNK * HID);
        tdm_load_tile_2d(pw + (size_t)(63 - j) * MAT_ELEMS + (size_t)n0g * HID + ko,
                         ldsB[t & 1], 256, CHUNK, HID);
        __builtin_amdgcn_s_wait_tensorcnt(2);   // stage s resident
      } else {
        __builtin_amdgcn_s_wait_tensorcnt(0);
      }
    }
    __syncthreads();
    const __bf16* tA = &sA[buf][0];
    const __bf16* tB = &sB[buf][0];
#pragma unroll
    for (int kb = 0; kb < 256; kb += 32) {
      bf16x16 a = load_frag_a(tA, 256, m0, kb);           // reused for both N tiles
      acc0 = wmma_bf16(a, load_frag_bT(tB, 256, nl0, kb), acc0);
      acc1 = wmma_bf16(a, load_frag_bT(tB, 256, nl0 + 16, kb), acc1);
    }
    __syncthreads();
  }
  int lane = threadIdx.x & 31;
  int n = n0g + nl0 + (lane & 15);
  int mbase = m0 + ((lane >> 4) << 3);
#pragma unroll
  for (int r = 0; r < 8; ++r) {
    Y[(size_t)(mbase + r) * HID + n] = acc0[r];
    Y[(size_t)(mbase + r) * HID + n + 16] = acc1[r];
  }
}

// ---------------------------------------------------------------------------
// Kernel 5: 64-step scan  h <- (Wh^64) h + Y_c, h in LDS, A = pw[64] (bf16).
// ---------------------------------------------------------------------------
__global__ __launch_bounds__(512) void k_scan(
    const __bf16* __restrict__ pw, const float* __restrict__ Y,
    float* __restrict__ hfin) {
  __shared__ float h[HID];
  int tid = threadIdx.x;
  h[tid] = 0.f;
  __syncthreads();
  const __bf16* Arow = pw + (size_t)64 * MAT_ELEMS + (size_t)tid * HID;
  for (int c = 0; c < NCHUNK; ++c) {
    float acc = Y[c * HID + tid];
#pragma unroll 8
    for (int k = 0; k < HID; k += 8) {
      bf16x8 av = *(const bf16x8*)(Arow + k);
#pragma unroll
      for (int j = 0; j < 8; ++j) acc += (float)av[j] * h[k + j];
    }
    __syncthreads();
    h[tid] = acc;
    __syncthreads();
  }
  hfin[tid] = h[tid];
}

// ---------------------------------------------------------------------------
// Kernel 6: logits = h @ Wo.T + bo, then log_softmax (5 outputs). One wave32.
// ---------------------------------------------------------------------------
__global__ void k_logits(const float* __restrict__ hfin, const float* __restrict__ Wo,
                         const float* __restrict__ bo, float* __restrict__ out) {
  int lane = threadIdx.x;
  float p[OUT_DIM];
#pragma unroll
  for (int o = 0; o < OUT_DIM; ++o) p[o] = 0.f;
  for (int k = lane; k < HID; k += 32) {
    float hv = hfin[k];
#pragma unroll
    for (int o = 0; o < OUT_DIM; ++o) p[o] += hv * Wo[(size_t)o * HID + k];
  }
#pragma unroll
  for (int o = 0; o < OUT_DIM; ++o)
    for (int off = 16; off > 0; off >>= 1) p[o] += __shfl_xor(p[o], off, 32);
  if (lane == 0) {
    float lg[OUT_DIM], mx = -3.0e38f;
#pragma unroll
    for (int o = 0; o < OUT_DIM; ++o) { lg[o] = p[o] + bo[o]; mx = fmaxf(mx, lg[o]); }
    float s = 0.f;
#pragma unroll
    for (int o = 0; o < OUT_DIM; ++o) s += expf(lg[o] - mx);
    float lse = mx + logf(s);
#pragma unroll
    for (int o = 0; o < OUT_DIM; ++o) out[o] = lg[o] - lse;
  }
}

// ---------------------------------------------------------------------------
extern "C" void kernel_launch(void* const* d_in, const int* in_sizes, int n_in,
                              void* d_out, int out_size, void* d_ws, size_t ws_size,
                              hipStream_t stream) {
  (void)in_sizes; (void)n_in; (void)out_size; (void)ws_size;
  const int*   tokens = (const int*)d_in[0];
  const float* emb    = (const float*)d_in[1];
  const float* Wi     = (const float*)d_in[2];
  const float* bi     = (const float*)d_in[3];
  const float* Wo     = (const float*)d_in[4];
  const float* bo     = (const float*)d_in[5];
  float* out = (float*)d_out;

  char* ws = (char*)d_ws;
  __bf16* xproj = (__bf16*)ws;
  size_t off = (size_t)SEQ_LEN * HID * 2;
  __bf16* pw  = (__bf16*)(ws + off);  off += 65 * MAT_BYTES;   // P_0..P_64
  __bf16* pwT = (__bf16*)(ws + off);  off += 33 * MAT_BYTES;   // P_0..P_32 transposed
  float*  Y   = (float*)(ws + off);   off += (size_t)NCHUNK * HID * 4;
  float*  hfin = (float*)(ws + off);

  k_init_pows<<<MAT_ELEMS / 256, 256, 0, stream>>>(Wi, pw, pwT);
  k_embed_xproj<<<(SEQ_LEN / 16) * (HID / 16) / 8, 256, 0, stream>>>(
      tokens, emb, Wi, bi, xproj);

  // log-depth power chain: have = 1 -> 2 -> 4 -> 8 -> 16 -> 32 -> 64
  int have = 1;
  while (have < 64) {
    int count = (have < 64 - have) ? have : (64 - have);
    dim3 grid(128, (unsigned)count);
    k_power_gemm<<<grid, 256, 0, stream>>>(pw, pwT, have);
    have += count;
  }

  k_chunk_gemm<<<HID / 64, 256, 0, stream>>>(xproj, pw, Y);
  k_scan<<<1, 512, 0, stream>>>(pw, Y, hfin);
  k_logits<<<1, 32, 0, stream>>>(hfin, Wo, bo, out);
}
